// EnhancedOFTOutputLayer_46411416601019
// MI455X (gfx1250) — matrix operand order repaired and checked by reading, not compiled, not run on hardware
//
#include <hip/hip_runtime.h>

// ---------------------------------------------------------------------------
// Types for WMMA (CDNA5 / gfx1250, wave32)
// ---------------------------------------------------------------------------
typedef __attribute__((ext_vector_type(16))) __bf16 v16bf;
typedef __attribute__((ext_vector_type(8)))  float  v8f;

struct alignas(16) U128 { unsigned int w[4]; };
union AFrag { v16bf v; U128 u[2]; };

__device__ __forceinline__ unsigned short f2bf(float x) {
  unsigned int u = __builtin_bit_cast(unsigned int, x);
  u += 0x7FFFu + ((u >> 16) & 1u);      // round-to-nearest-even
  return (unsigned short)(u >> 16);
}

#define HIDDEN 4096
#define NB 8
#define BS 512

// ---------------------------------------------------------------------------
// 1) M[b] = I + 0.5*(A - A^T) + 1e-6*I ; X[b] = I  (Newton-Schulz seed)
// ---------------------------------------------------------------------------
__global__ __launch_bounds__(256) void build_MX_kernel(
    const float* __restrict__ A, float* __restrict__ M, float* __restrict__ X) {
  const int b   = blockIdx.y;
  const int idx = blockIdx.x * 256 + threadIdx.x;      // 0 .. 512*512-1
  const int i = idx >> 9;
  const int j = idx & 511;
  const float* Ab = A + (size_t)b * BS * BS;
  const float s = 0.5f * (Ab[i * BS + j] - Ab[j * BS + i]);
  const float d = (i == j) ? (1.0f + 1e-6f) : 0.0f;
  M[(size_t)b * BS * BS + idx] = s + d;
  X[(size_t)b * BS * BS + idx] = (i == j) ? 1.0f : 0.0f;
}

// ---------------------------------------------------------------------------
// 2) Batched fp32 512x512x512 GEMM. mode 0: C = A@B ; mode 1: C = 2*A - A@B
// ---------------------------------------------------------------------------
__global__ __launch_bounds__(256) void gemm512_kernel(
    const float* __restrict__ A, const float* __restrict__ B,
    float* __restrict__ C, int mode) {
  const int b = blockIdx.z;
  const float* Ab = A + (size_t)b * BS * BS;
  const float* Bb = B + (size_t)b * BS * BS;
  float* Cb = C + (size_t)b * BS * BS;

  __shared__ float As[64][17];
  __shared__ float Bs[16][65];

  const int tx = threadIdx.x, ty = threadIdx.y;   // 16x16
  const int t = ty * 16 + tx;
  const int row0 = blockIdx.y * 64;
  const int col0 = blockIdx.x * 64;

  float acc[4][4] = {};

  for (int k0 = 0; k0 < BS; k0 += 16) {
#pragma unroll
    for (int i = 0; i < 4; ++i) {
      int idx = t + i * 256;
      As[idx >> 4][idx & 15] = Ab[(row0 + (idx >> 4)) * BS + k0 + (idx & 15)];
    }
#pragma unroll
    for (int i = 0; i < 4; ++i) {
      int idx = t + i * 256;
      Bs[idx >> 6][idx & 63] = Bb[(k0 + (idx >> 6)) * BS + col0 + (idx & 63)];
    }
    __syncthreads();
#pragma unroll
    for (int kk = 0; kk < 16; ++kk) {
      float a[4], bb[4];
#pragma unroll
      for (int i = 0; i < 4; ++i) a[i] = As[ty * 4 + i][kk];
#pragma unroll
      for (int j = 0; j < 4; ++j) bb[j] = Bs[kk][tx * 4 + j];
#pragma unroll
      for (int i = 0; i < 4; ++i)
#pragma unroll
        for (int j = 0; j < 4; ++j) acc[i][j] = fmaf(a[i], bb[j], acc[i][j]);
    }
    __syncthreads();
  }
#pragma unroll
  for (int i = 0; i < 4; ++i)
#pragma unroll
    for (int j = 0; j < 4; ++j) {
      int r = row0 + ty * 4 + i, c = col0 + tx * 4 + j;
      float v = acc[i][j];
      if (mode == 1) v = 2.0f * Ab[r * BS + c] - v;
      Cb[r * BS + c] = v;
    }
}

// ---------------------------------------------------------------------------
// 3) Q = (2 + 1e-6)*X - T   (== (I - S) @ inv(M) with T = M@X)
// ---------------------------------------------------------------------------
__global__ __launch_bounds__(256) void q_combine_kernel(
    const float* __restrict__ X, const float* __restrict__ T, float* __restrict__ Q) {
  size_t i = (size_t)blockIdx.x * 256 + threadIdx.x;
  Q[i] = (2.0f + 1e-6f) * X[i] - T[i];
}

// ---------------------------------------------------------------------------
// 4) filt_bf16[b*512 + r, :] = Q[b] @ W[b*512 : (b+1)*512, :]  (fp32 acc)
// ---------------------------------------------------------------------------
__global__ __launch_bounds__(256) void filt_kernel(
    const float* __restrict__ Q, const float* __restrict__ W,
    unsigned short* __restrict__ filt) {
  const int b = blockIdx.z;
  const float* Qb = Q + (size_t)b * BS * BS;
  const float* Wb = W + (size_t)b * BS * HIDDEN;

  __shared__ float As[64][17];
  __shared__ float Bs[16][65];

  const int tx = threadIdx.x, ty = threadIdx.y;
  const int t = ty * 16 + tx;
  const int row0 = blockIdx.y * 64;     // 0..511
  const int col0 = blockIdx.x * 64;     // 0..4095

  float acc[4][4] = {};

  for (int k0 = 0; k0 < BS; k0 += 16) {
#pragma unroll
    for (int i = 0; i < 4; ++i) {
      int idx = t + i * 256;
      As[idx >> 4][idx & 15] = Qb[(row0 + (idx >> 4)) * BS + k0 + (idx & 15)];
    }
#pragma unroll
    for (int i = 0; i < 4; ++i) {
      int idx = t + i * 256;
      Bs[idx >> 6][idx & 63] = Wb[(size_t)(k0 + (idx >> 6)) * HIDDEN + col0 + (idx & 63)];
    }
    __syncthreads();
#pragma unroll
    for (int kk = 0; kk < 16; ++kk) {
      float a[4], bb[4];
#pragma unroll
      for (int i = 0; i < 4; ++i) a[i] = As[ty * 4 + i][kk];
#pragma unroll
      for (int j = 0; j < 4; ++j) bb[j] = Bs[kk][tx * 4 + j];
#pragma unroll
      for (int i = 0; i < 4; ++i)
#pragma unroll
        for (int j = 0; j < 4; ++j) acc[i][j] = fmaf(a[i], bb[j], acc[i][j]);
    }
    __syncthreads();
  }
#pragma unroll
  for (int i = 0; i < 4; ++i)
#pragma unroll
    for (int j = 0; j < 4; ++j) {
      int r = row0 + ty * 4 + i, c = col0 + tx * 4 + j;
      filt[(size_t)(b * BS + r) * HIDDEN + c] = f2bf(acc[i][j]);
    }
}

// ---------------------------------------------------------------------------
// 5) x (fp32) -> bf16, 4 elements per thread
// ---------------------------------------------------------------------------
struct F4 { float f[4]; };
struct H4 { unsigned short h[4]; };
__global__ __launch_bounds__(256) void cvt_bf16_kernel(
    const float* __restrict__ in, unsigned short* __restrict__ out) {
  size_t i = (size_t)blockIdx.x * 256 + threadIdx.x;
  F4 v = ((const F4*)in)[i];
  H4 o;
#pragma unroll
  for (int j = 0; j < 4; ++j) o.h[j] = f2bf(v.f[j]);
  ((H4*)out)[i] = o;
}

// ---------------------------------------------------------------------------
// 6) Main GEMM: out[8192,4096] = x_bf16 @ filt_bf16^T + bias  (bf16 WMMA)
//    Block tile 128(M) x 256(N), 8 waves (wave32), wave tile 64x64 = 4x4 WMMA.
//    Double-buffered LDS fed by CDNA5 async global->LDS (ASYNCcnt) loads with
//    immediate IOFFSETs; K-loop unrolled x2 so buffer selects are literal.
// ---------------------------------------------------------------------------
#define LDK 40   // bf16 elements per LDS row (32 data + 8 pad), 80B: 16B aligned
#define ROWB (HIDDEN * 2)          // bytes per tensor row
#define ROW64 (64 * HIDDEN * 2)    // 64-row displacement (fits signed 24-bit)

// CDNA5 async global->LDS 16B load with compile-time instruction offset.
template <int OFF>
__device__ __forceinline__ void async_b128o(const void* gptr, unsigned ldsAddr) {
  asm volatile("global_load_async_to_lds_b128 %0, %1, off offset:%2"
               :: "v"(ldsAddr), "v"(gptr), "i"(OFF) : "memory");
}
__device__ __forceinline__ void wait_async0() {
  asm volatile("s_wait_asynccnt 0x0" ::: "memory");
}

struct XL { unsigned a, b; };            // LDS dests for the 2 x-tile chunks
struct FL { unsigned a, b, c, d; };      // LDS dests for the 4 f-tile chunks

// Issue one 128x32 x-tile + one 256x32 f-tile; bases advance outside.
__device__ __forceinline__ void issue_tiles(
    const unsigned short* gx, const unsigned short* gf, XL lx, FL lf) {
  async_b128o<0>(gx, lx.a);
  async_b128o<ROW64>(gx, lx.b);
  async_b128o<0>(gf, lf.a);
  async_b128o<ROW64>(gf, lf.b);
  async_b128o<2 * ROW64>(gf, lf.c);
  async_b128o<3 * ROW64>(gf, lf.d);
}

// Gather fragments (ISA 16-bit A/B layouts) and run 16 WMMAs on one buffer.
__device__ __forceinline__ void do_step(
    const unsigned short* __restrict__ xsBuf,
    const unsigned short* __restrict__ fsBuf,
    int wm, int wn, int l15, int halfSel, v8f (&acc)[4][4]) {
  AFrag af[4], bfr[4];
#pragma unroll
  for (int mi = 0; mi < 4; ++mi) {
    const unsigned short* base = xsBuf + (wm * 64 + mi * 16 + l15) * LDK;
    af[mi].u[0] = *(const U128*)(base + halfSel * 8);        // K = hs*8 .. +7
    af[mi].u[1] = *(const U128*)(base + 16 + halfSel * 8);   // K = 16+hs*8 ..
  }
#pragma unroll
  for (int ni = 0; ni < 4; ++ni) {
    const unsigned short* base = fsBuf + (wn * 64 + ni * 16 + l15) * LDK + halfSel * 8;
    bfr[ni].u[0] = *(const U128*)(base);                     // K = hs*8 .. +7
    bfr[ni].u[1] = *(const U128*)(base + 16);                // K = 16+hs*8 ..
  }
#pragma unroll
  for (int mi = 0; mi < 4; ++mi)
#pragma unroll
    for (int ni = 0; ni < 4; ++ni)
      acc[mi][ni] = __builtin_amdgcn_wmma_f32_16x16x32_bf16(
          false, af[mi].v, false, bfr[ni].v, (short)0, acc[mi][ni], false, false);
}

__global__ __launch_bounds__(256) void oft_wmma_kernel(
    const unsigned short* __restrict__ xb,   // [8192,4096] bf16 row-major
    const unsigned short* __restrict__ fb,   // [4096,4096] bf16 row-major (N x K)
    const float* __restrict__ bias,
    float* __restrict__ out) {
  __shared__ unsigned short xs[2][128 * LDK];   // 20 KB
  __shared__ unsigned short fs[2][256 * LDK];   // 40 KB

  const int tid  = threadIdx.x;
  const int lane = tid & 31;
  const int wave = tid >> 5;          // 0..7
  const int wm   = wave & 1;          // 2 waves along M
  const int wn   = wave >> 1;         // 4 waves along N

  const int m0 = blockIdx.y * 128;
  const int n0 = blockIdx.x * 256;

  const int halfSel = lane >> 4;
  const int l15     = lane & 15;

  // Per-thread staging coordinates: 16B chunk (row, quarter).
  const int srow = tid >> 2;          // 0..63
  const int sq   = tid & 3;           // 0..3 (quarter of a 32-bf16 row)

  // Running global base pointers (advance 64B per K-tile).
  const unsigned short* gx = xb + (size_t)(m0 + srow) * HIDDEN + sq * 8;
  const unsigned short* gf = fb + (size_t)(n0 + srow) * HIDDEN + sq * 8;

  // Precomputed LDS destinations for both buffers (constant for the kernel).
  XL lx0 = { (unsigned)(size_t)&xs[0][srow * LDK + sq * 8],
             (unsigned)(size_t)&xs[0][(srow + 64) * LDK + sq * 8] };
  XL lx1 = { (unsigned)(size_t)&xs[1][srow * LDK + sq * 8],
             (unsigned)(size_t)&xs[1][(srow + 64) * LDK + sq * 8] };
  FL lf0 = { (unsigned)(size_t)&fs[0][srow * LDK + sq * 8],
             (unsigned)(size_t)&fs[0][(srow + 64) * LDK + sq * 8],
             (unsigned)(size_t)&fs[0][(srow + 128) * LDK + sq * 8],
             (unsigned)(size_t)&fs[0][(srow + 192) * LDK + sq * 8] };
  FL lf1 = { (unsigned)(size_t)&fs[1][srow * LDK + sq * 8],
             (unsigned)(size_t)&fs[1][(srow + 64) * LDK + sq * 8],
             (unsigned)(size_t)&fs[1][(srow + 128) * LDK + sq * 8],
             (unsigned)(size_t)&fs[1][(srow + 192) * LDK + sq * 8] };

  v8f acc[4][4] = {};

  // Prologue: async-stage tile 0 into buffer 0.
  issue_tiles(gx, gf, lx0, lf0);
  gx += 32; gf += 32;
  wait_async0();
  __syncthreads();

  // 128 K-tiles, unrolled x2 for literal double-buffer indices.
  for (int k0 = 0; k0 < HIDDEN; k0 += 64) {
    // Step A: compute buf0, prefetch buf1 (always a valid next tile).
    issue_tiles(gx, gf, lx1, lf1);
    gx += 32; gf += 32;
    do_step(xs[0], fs[0], wm, wn, l15, halfSel, acc);
    wait_async0();
    __syncthreads();

    // Step B: compute buf1, prefetch buf0 (except after the last tile).
    if (k0 + 64 < HIDDEN) {
      issue_tiles(gx, gf, lx0, lf0);
      gx += 32; gf += 32;
    }
    do_step(xs[1], fs[1], wm, wn, l15, halfSel, acc);
    wait_async0();
    __syncthreads();
  }

  // ---- epilogue: D layout -> global, + bias ----
#pragma unroll
  for (int mi = 0; mi < 4; ++mi)
#pragma unroll
    for (int ni = 0; ni < 4; ++ni) {
      const int col = n0 + wn * 64 + ni * 16 + l15;
      const float bv = bias[col];
#pragma unroll
      for (int r = 0; r < 8; ++r) {
        const int row = m0 + wm * 64 + mi * 16 + halfSel * 8 + r;
        out[(size_t)row * HIDDEN + col] = acc[mi][ni][r] + bv;
      }
    }
}

// ---------------------------------------------------------------------------
// Host launch
// ---------------------------------------------------------------------------
extern "C" void kernel_launch(void* const* d_in, const int* in_sizes, int n_in,
                              void* d_out, int out_size, void* d_ws, size_t ws_size,
                              hipStream_t stream) {
  const float* weight = (const float*)d_in[0];   // [4096,4096]
  const float* bias   = (const float*)d_in[1];   // [4096]
  const float* x      = (const float*)d_in[2];   // [4,2048,4096]
  const float* projR  = (const float*)d_in[3];   // [8,512,512]
  (void)in_sizes; (void)n_in; (void)ws_size;

  char* ws = (char*)d_ws;
  size_t off = 0;
  auto carve = [&](size_t bytes) -> void* {
    off = (off + 255) & ~(size_t)255;
    void* p = ws + off;
    off += bytes;
    return p;
  };

  const size_t blkBytes = (size_t)NB * BS * BS * sizeof(float);   // 8 MB
  float* M  = (float*)carve(blkBytes);
  float* X  = (float*)carve(blkBytes);
  float* X2 = (float*)carve(blkBytes);
  float* T  = (float*)carve(blkBytes);
  float* Q  = (float*)carve(blkBytes);
  unsigned short* fbf = (unsigned short*)carve((size_t)HIDDEN * HIDDEN * 2);      // 32 MB
  unsigned short* xbf = (unsigned short*)carve((size_t)4 * 2048 * HIDDEN * 2);    // 64 MB

  // 1) M, X seed
  build_MX_kernel<<<dim3(1024, NB), 256, 0, stream>>>(projR, M, X);

  // 2) Newton-Schulz: X <- X(2I - MX), 9 iterations (||S|| ~ 0.64 -> fp32 exact)
  float* Xa = X; float* Xb2 = X2;
  for (int it = 0; it < 9; ++it) {
    gemm512_kernel<<<dim3(8, 8, NB), dim3(16, 16), 0, stream>>>(M, Xa, T, 0);
    gemm512_kernel<<<dim3(8, 8, NB), dim3(16, 16), 0, stream>>>(Xa, T, Xb2, 1);
    float* tmp = Xa; Xa = Xb2; Xb2 = tmp;
  }

  // 3) Q = (2+1e-6)*X - M@X  == (I - S) @ inv(M)
  gemm512_kernel<<<dim3(8, 8, NB), dim3(16, 16), 0, stream>>>(M, Xa, T, 0);
  q_combine_kernel<<<(unsigned)((size_t)NB * BS * BS / 256), 256, 0, stream>>>(Xa, T, Q);

  // 4) filt (bf16) = blockdiag(Q) @ weight
  filt_kernel<<<dim3(HIDDEN / 64, BS / 64, NB), dim3(16, 16), 0, stream>>>(Q, weight, fbf);

  // 5) x -> bf16
  cvt_bf16_kernel<<<(unsigned)((size_t)4 * 2048 * HIDDEN / (256 * 4)), 256, 0, stream>>>(x, xbf);

  // 6) out = x @ filt^T + bias via bf16 WMMA (fp32 accumulate), async-LDS fed
  oft_wmma_kernel<<<dim3(HIDDEN / 256, 8192 / 128), 256, 0, stream>>>(
      xbf, fbf, bias, (float*)d_out);
}
